// CaptioningRNN_36893769073190
// MI455X (gfx1250) — matrix operand order repaired and checked by reading, not compiled, not run on hardware
//
#include <hip/hip_runtime.h>
#include <hip/hip_bf16.h>

// ---------------------------------------------------------------------------
// CaptioningRNN (attention LSTM) forward loss for MI455X / gfx1250.
// Mixed precision: bf16 WMMA (v_wmma_f32_16x16x32_bf16) with f32 accumulate
// for all GEMMs; f32 scalar math for softmax / gates / logsumexp.
// ---------------------------------------------------------------------------

#define NB    256      // batch
#define TT    32       // time steps (Tp1-1)
#define VV    10000    // vocab
#define DF    1280     // D_FEAT
#define HH    512      // hidden
#define WDIM  256      // word dim
#define PP    16       // 4x4 spatial positions
#define KCAT  (WDIM + HH + HH)   // 1280
#define G4H   (4 * HH)           // 2048

typedef __attribute__((ext_vector_type(4)))  unsigned int v4u;
typedef __attribute__((ext_vector_type(16))) __bf16       v16bf;
typedef __attribute__((ext_vector_type(8)))  float        v8f;

union Frag { v4u u[2]; v16bf v; };

// ---------------------------------------------------------------------------
// small helpers
// ---------------------------------------------------------------------------
__global__ void k_zero_out(float* out) { out[0] = 0.0f; }

// dst[Nn][K] (bf16, row-major) = transpose of src[K][Nn] (f32)
__global__ void k_t2bf(__bf16* __restrict__ dst, const float* __restrict__ src,
                       int K, int Nn)
{
  size_t total = (size_t)K * Nn;
  for (size_t i = (size_t)blockIdx.x * blockDim.x + threadIdx.x; i < total;
       i += (size_t)gridDim.x * blockDim.x) {
    size_t nn = i / K, k = i % K;
    dst[i] = (__bf16)src[k * (size_t)Nn + nn];
  }
}

// WcatT[2048][1280] = transpose of [Wx;Wh;Wattn] (stacked along K)
__global__ void k_wcatT(__bf16* __restrict__ dst,
                        const float* __restrict__ Wx,
                        const float* __restrict__ Wh,
                        const float* __restrict__ Wattn)
{
  size_t total = (size_t)G4H * KCAT;
  for (size_t i = (size_t)blockIdx.x * blockDim.x + threadIdx.x; i < total;
       i += (size_t)gridDim.x * blockDim.x) {
    size_t nc = i / KCAT, k = i % KCAT;
    float v;
    if (k < WDIM)            v = Wx[k * (size_t)G4H + nc];
    else if (k < WDIM + HH)  v = Wh[(k - WDIM) * (size_t)G4H + nc];
    else                     v = Wattn[(k - WDIM - HH) * (size_t)G4H + nc];
    dst[i] = (__bf16)v;
  }
}

// Xp[(n*16+p)][c] (bf16) = feats[n][c][p]  (feats: N x 1280 x 16)
__global__ void k_feats_xp(__bf16* __restrict__ dst, const float* __restrict__ feats)
{
  size_t total = (size_t)NB * PP * DF;
  for (size_t i = (size_t)blockIdx.x * blockDim.x + threadIdx.x; i < total;
       i += (size_t)gridDim.x * blockDim.x) {
    size_t row = i / DF, c = i % DF;
    size_t n = row >> 4, p = row & 15;
    dst[i] = (__bf16)feats[(n * DF + c) * PP + p];
  }
}

// ---------------------------------------------------------------------------
// Generic bf16 GEMM:  C[M][Nn] (f32) = A[M][K] (bf16, rm) x Bt[Nn][K]^T + bias
// Wave tile: 32(M) x 64(N).  Workgroup (256 thr = 8 waves): 64(M) x 256(N).
// K stepped by 32 (one v_wmma_f32_16x16x32_bf16 per frag pair).
// Prefetches are unconditional: speculative global_prefetch_b8 drops
// translation failures silently (ISA 10.5), so running past the panel end
// is harmless and keeps the inner loop branch-free.
// ---------------------------------------------------------------------------
__global__ void k_gemm_bf16(float* __restrict__ C,
                            const __bf16* __restrict__ A,
                            const __bf16* __restrict__ Bt,
                            const float* __restrict__ bias,
                            int M, int Nn, int K)
{
  const int lane = threadIdx.x & 31;
  const int wave = threadIdx.x >> 5;
  const int m0 = blockIdx.y * 64 + (wave & 1) * 32;
  const int n0 = blockIdx.x * 256 + (wave >> 1) * 64;
  const int r   = lane & 15;
  const int s0a = (lane < 16) ? 0 : 16;   // A: K 0..15 in lanes 0-15, 16..31 hi
  const int s0b = (lane < 16) ? 0 : 16;   // B: K 0..15 in lanes 0-15, 16..31 hi

  const v8f vz = {0.f, 0.f, 0.f, 0.f, 0.f, 0.f, 0.f, 0.f};
  v8f acc[2][4];
  for (int i = 0; i < 2; ++i)
    for (int j = 0; j < 4; ++j) acc[i][j] = vz;

  for (int k0 = 0; k0 < K; k0 += 32) {
    Frag a[2], b[4];
#pragma unroll
    for (int i = 0; i < 2; ++i) {
      const __bf16* p = A + (size_t)(m0 + 16 * i + r) * K + k0 + s0a;
      a[i].u[0] = *(const v4u*)p;
      a[i].u[1] = *(const v4u*)(p + 8);
    }
#pragma unroll
    for (int j = 0; j < 4; ++j) {
      const __bf16* p = Bt + (size_t)(n0 + 16 * j + r) * K + k0 + s0b;
      b[j].u[0] = *(const v4u*)p;
      b[j].u[1] = *(const v4u*)(p + 8);
      __builtin_prefetch(p + 32, 0, 3);   // speculative global_prefetch_b8
    }
#pragma unroll
    for (int i = 0; i < 2; ++i)
#pragma unroll
      for (int j = 0; j < 4; ++j)
        acc[i][j] = __builtin_amdgcn_wmma_f32_16x16x32_bf16(
            false, a[i].v, false, b[j].v, (short)0, acc[i][j], false, false);
  }

  const int rowHalf = (lane < 16) ? 0 : 8;
  const int col = lane & 15;
#pragma unroll
  for (int i = 0; i < 2; ++i)
#pragma unroll
    for (int j = 0; j < 4; ++j) {
      int nc = n0 + 16 * j + col;
      float bvv = bias ? bias[nc] : 0.0f;
#pragma unroll
      for (int v = 0; v < 8; ++v) {
        int mr = m0 + 16 * i + v + rowHalf;
        C[(size_t)mr * Nn + nc] = acc[i][j][v] + bvv;
      }
    }
}

// ---------------------------------------------------------------------------
// h0 = mean over 16 positions of A; also initializes c = h0
// A layout: [(n*16+p)][h] f32
// ---------------------------------------------------------------------------
__global__ void k_h0(float* __restrict__ h, float* __restrict__ c,
                     const float* __restrict__ A)
{
  int idx = blockIdx.x * 256 + threadIdx.x;   // N*H
  int n = idx >> 9, hh = idx & 511;
  float s = 0.f;
#pragma unroll
  for (int p = 0; p < PP; ++p) s += A[((size_t)(n * PP + p)) * HH + hh];
  s *= (1.0f / 16.0f);
  h[idx] = s;
  c[idx] = s;
}

// ---------------------------------------------------------------------------
// Per-step: attention (scores->softmax->attn) and build U = [x_t | h | attn]
// One workgroup (256 threads) per sample n.
// ---------------------------------------------------------------------------
__global__ void k_attn_u(__bf16* __restrict__ U,
                         const float* __restrict__ A,
                         const float* __restrict__ h,
                         const float* __restrict__ Wemb,
                         const int* __restrict__ captions, int t)
{
  __shared__ float hloc[HH];
  __shared__ float sc[PP][16];
  __shared__ float w[PP];
  const int n = blockIdx.x, tid = threadIdx.x;

  hloc[tid]        = h[(size_t)n * HH + tid];
  hloc[tid + 256]  = h[(size_t)n * HH + 256 + tid];
  __syncthreads();

  // scores[p] = (1/sqrt(H)) * sum_h h[n,h] * A[(n,p),h]
  const int p = tid >> 4, j = tid & 15;
  const float* Ar = A + ((size_t)(n * PP + p)) * HH;
  float part = 0.f;
  for (int k = j; k < HH; k += 16) part += hloc[k] * Ar[k];
  sc[p][j] = part;
  __syncthreads();
  if (tid < PP) {
    float s = 0.f;
    for (int q = 0; q < 16; ++q) s += sc[tid][q];
    sc[tid][0] = s * 0.04419417382415922f;   // 1/sqrt(512)
  }
  __syncthreads();
  if (tid == 0) {
    float m = -3.0e38f;
    for (int q = 0; q < PP; ++q) m = fmaxf(m, sc[q][0]);
    float e[PP], s = 0.f;
    for (int q = 0; q < PP; ++q) { e[q] = __expf(sc[q][0] - m); s += e[q]; }
    float inv = 1.f / s;
    for (int q = 0; q < PP; ++q) w[q] = e[q] * inv;
  }
  __syncthreads();

  __bf16* Un = U + (size_t)n * KCAT;
  // attn[h] = sum_p A[(n,p),h] * w[p];  also copy h into U (bf16)
  for (int hh = tid; hh < HH; hh += 256) {
    float s = 0.f;
#pragma unroll
    for (int q = 0; q < PP; ++q) s += A[((size_t)(n * PP + q)) * HH + hh] * w[q];
    Un[WDIM + HH + hh] = (__bf16)s;        // attn block
    Un[WDIM + hh]      = (__bf16)hloc[hh]; // h block
  }
  // x_t = W_embed[captions[n,t]]
  const int cap = captions[n * 33 + t];
  Un[tid] = (__bf16)Wemb[(size_t)cap * WDIM + tid];
}

// ---------------------------------------------------------------------------
// LSTM pointwise: gates from a[N][4H], update (h,c), stash h as bf16 in hs
// ---------------------------------------------------------------------------
__global__ void k_lstm(float* __restrict__ h, float* __restrict__ c,
                       __bf16* __restrict__ hs, const float* __restrict__ a, int t)
{
  int idx = blockIdx.x * 256 + threadIdx.x;  // N*H
  int n = idx >> 9, hh = idx & 511;
  const float* an = a + (size_t)n * G4H;
  float gi = 1.f / (1.f + __expf(-an[hh]));
  float gf = 1.f / (1.f + __expf(-an[HH + hh]));
  float go = 1.f / (1.f + __expf(-an[2 * HH + hh]));
  float gg = tanhf(an[3 * HH + hh]);
  float cn = gf * c[idx] + gi * gg;
  float hn = go * tanhf(cn);
  c[idx] = cn;
  h[idx] = hn;
  hs[((size_t)n * TT + t) * HH + hh] = (__bf16)hn;
}

// ---------------------------------------------------------------------------
// Fused vocab GEMM + online logsumexp + masked NLL reduction.
// One workgroup per 16-row tile of hs (8192/16 = 512 WGs, 8 waves each).
// Waves sweep 625 v-tiles of 16 columns; per-lane online (max,sumexp).
// ---------------------------------------------------------------------------
__global__ void k_vocab_loss(float* __restrict__ out,
                             const __bf16* __restrict__ hs,
                             const __bf16* __restrict__ WvT,   // [V][H] bf16
                             const float* __restrict__ Wv,     // [H][V] f32
                             const float* __restrict__ bvoc,   // [V]
                             const int* __restrict__ captions)
{
  __shared__ __align__(16) __bf16 lds_h[16 * HH];   // 16 KB
  __shared__ float l_max[8][32][8];
  __shared__ float l_sum[8][32][8];
  __shared__ float red[256];
  __shared__ float row_lse[16];

  const int tid = threadIdx.x, lane = tid & 31, wave = tid >> 5;
  const int r0 = blockIdx.x * 16;

  // stage 16 contiguous rows of hs (bf16) into LDS
  {
    const v4u* src = (const v4u*)(hs + (size_t)r0 * HH);
    v4u* dst = (v4u*)lds_h;
    for (int i = tid; i < 16 * HH / 8; i += 256) dst[i] = src[i];
  }
  __syncthreads();

  const int r   = lane & 15;
  const int s0a = (lane < 16) ? 0 : 16;
  const int s0b = (lane < 16) ? 0 : 16;

  float rmax[8], rsum[8];
#pragma unroll
  for (int v = 0; v < 8; ++v) { rmax[v] = -3.0e38f; rsum[v] = 0.f; }

  for (int vt = wave; vt < VV / 16; vt += 8) {
    const int v0 = vt * 16;
    v8f acc = {0.f, 0.f, 0.f, 0.f, 0.f, 0.f, 0.f, 0.f};
#pragma unroll 4
    for (int k0 = 0; k0 < HH; k0 += 32) {
      Frag a, b;
      const __bf16* pa = lds_h + r * HH + k0 + s0a;
      a.u[0] = *(const v4u*)pa;
      a.u[1] = *(const v4u*)(pa + 8);
      const __bf16* pb = WvT + (size_t)(v0 + r) * HH + k0 + s0b;
      b.u[0] = *(const v4u*)pb;
      b.u[1] = *(const v4u*)(pb + 8);
      acc = __builtin_amdgcn_wmma_f32_16x16x32_bf16(
          false, a.v, false, b.v, (short)0, acc, false, false);
    }
    const float biasv = bvoc[v0 + (lane & 15)];
#pragma unroll
    for (int v = 0; v < 8; ++v) {
      float s  = acc[v] + biasv;
      float m  = rmax[v];
      float nm = fmaxf(m, s);
      rsum[v]  = rsum[v] * __expf(m - nm) + __expf(s - nm);
      rmax[v]  = nm;
    }
  }
#pragma unroll
  for (int v = 0; v < 8; ++v) {
    l_max[wave][lane][v] = rmax[v];
    l_sum[wave][lane][v] = rsum[v];
  }
  __syncthreads();

  // combine online stats: one thread per row
  if (tid < 16) {
    const int half = tid >> 3;        // rows 0-7 live in lanes 0-15, 8-15 in 16-31
    const int v = tid & 7;
    float m = -3.0e38f, s = 0.f;
    for (int w2 = 0; w2 < 8; ++w2)
      for (int l = half * 16; l < half * 16 + 16; ++l) {
        float mm = l_max[w2][l][v], ss = l_sum[w2][l][v];
        float nm = fmaxf(m, mm);
        s = s * __expf(m - nm) + ss * __expf(mm - nm);
        m = nm;
      }
    row_lse[tid] = m + __logf(s);
  }
  __syncthreads();

  // target logits: 16 threads per row
  const int row = tid >> 4, j = tid & 15;
  const int g = r0 + row;
  const int n = g >> 5, t = g & 31;
  const int tgt = captions[n * 33 + t + 1];
  float part = 0.f;
  for (int k = j; k < HH; k += 16)
    part += (float)lds_h[row * HH + k] * Wv[(size_t)k * VV + tgt];
  red[tid] = part;
  __syncthreads();
  if (j == 0) {
    float s = 0.f;
    for (int q = 0; q < 16; ++q) s += red[(row << 4) + q];
    s += bvoc[tgt];
    if (tgt != 0)
      atomicAdd(out, (row_lse[row] - s) * (1.0f / 256.0f));
  }
}

// ---------------------------------------------------------------------------
// Host orchestration
// ---------------------------------------------------------------------------
extern "C" void kernel_launch(void* const* d_in, const int* in_sizes, int n_in,
                              void* d_out, int out_size, void* d_ws, size_t ws_size,
                              hipStream_t stream)
{
  const float* feats    = (const float*)d_in[0];
  const int*   captions = (const int*)d_in[1];
  const float* W_proj   = (const float*)d_in[2];
  const float* b_proj   = (const float*)d_in[3];
  const float* W_embed  = (const float*)d_in[4];
  const float* Wx       = (const float*)d_in[5];
  const float* Wh       = (const float*)d_in[6];
  const float* Wattn    = (const float*)d_in[7];
  const float* b        = (const float*)d_in[8];
  const float* W_vocab  = (const float*)d_in[9];
  const float* b_vocab  = (const float*)d_in[10];
  float* out = (float*)d_out;

  char* ws = (char*)d_ws;
  auto alloc = [&](size_t bytes) -> void* {
    void* p = ws;
    ws += (bytes + 255) & ~(size_t)255;
    return p;
  };

  __bf16* wcatT  = (__bf16*)alloc((size_t)G4H * KCAT * 2);      // [2048][1280]
  __bf16* wprojT = (__bf16*)alloc((size_t)HH * DF * 2);         // [512][1280]
  __bf16* wvocT  = (__bf16*)alloc((size_t)VV * HH * 2);         // [10000][512]
  __bf16* xp     = (__bf16*)alloc((size_t)NB * PP * DF * 2);    // [4096][1280]
  float*  A_f    = (float*) alloc((size_t)NB * PP * HH * 4);    // [4096][512]
  float*  h_st   = (float*) alloc((size_t)NB * HH * 4);
  float*  c_st   = (float*) alloc((size_t)NB * HH * 4);
  __bf16* U      = (__bf16*)alloc((size_t)NB * KCAT * 2);       // [256][1280]
  float*  a_f    = (float*) alloc((size_t)NB * G4H * 4);        // [256][2048]
  __bf16* hs     = (__bf16*)alloc((size_t)NB * TT * HH * 2);    // [8192][512]

  k_zero_out<<<1, 1, 0, stream>>>(out);

  // weight preparation (bf16, pre-transposed so WMMA B-frags are contiguous)
  {
    size_t tot;
    tot = (size_t)DF * HH;
    k_t2bf<<<(int)((tot + 255) / 256), 256, 0, stream>>>(wprojT, W_proj, DF, HH);
    tot = (size_t)HH * VV;
    k_t2bf<<<(int)((tot + 255) / 256), 256, 0, stream>>>(wvocT, W_vocab, HH, VV);
    tot = (size_t)G4H * KCAT;
    k_wcatT<<<(int)((tot + 255) / 256), 256, 0, stream>>>(wcatT, Wx, Wh, Wattn);
    tot = (size_t)NB * PP * DF;
    k_feats_xp<<<(int)((tot + 255) / 256), 256, 0, stream>>>(xp, feats);
  }

  // A = feats-proj GEMM: [4096 x 1280] x [1280 x 512] + b_proj
  {
    dim3 grid(HH / 256, (NB * PP) / 64);
    k_gemm_bf16<<<grid, 256, 0, stream>>>(A_f, xp, wprojT, b_proj,
                                          NB * PP, HH, DF);
  }

  // h0 = mean over positions; c0 = h0
  k_h0<<<(NB * HH) / 256, 256, 0, stream>>>(h_st, c_st, A_f);

  // sequential attention-LSTM
  for (int t = 0; t < TT; ++t) {
    k_attn_u<<<NB, 256, 0, stream>>>(U, A_f, h_st, W_embed, captions, t);
    dim3 grid(G4H / 256, NB / 64);
    k_gemm_bf16<<<grid, 256, 0, stream>>>(a_f, U, wcatT, b, NB, G4H, KCAT);
    k_lstm<<<(NB * HH) / 256, 256, 0, stream>>>(h_st, c_st, hs, a_f, t);
  }

  // fused vocab projection + log-softmax NLL
  k_vocab_loss<<<(NB * TT) / 16, 256, 0, stream>>>(out, hs, wvocT, W_vocab,
                                                   b_vocab, captions);
}
